// Attention_59863254172655
// MI455X (gfx1250) — matrix operand order repaired and checked by reading, not compiled, not run on hardware
//
#include <hip/hip_runtime.h>
#include <hip/hip_bf16.h>

#define NUM_HEADS 16
#define HIDDEN    1024
#define HEAD_DIM  64
#define SEQ       2048
#define BATCH     4

// Use CDNA5 async global->LDS staging (ASYNCcnt path) for K tiles.
#define USE_ASYNC_K 1

typedef __attribute__((ext_vector_type(16))) _Float16 v16h;
typedef __attribute__((ext_vector_type(8)))  float    v8f;

#define WMMA_F16(A, B, C) \
  __builtin_amdgcn_wmma_f32_16x16x32_f16(false, (A), false, (B), (short)0, (C), false, false)

__device__ __forceinline__ int lane_id() { return (int)(threadIdx.x & 31u); }

// LDS byte offset of a __shared__ object (ptrtoint of addrspace(3) pointer)
__device__ __forceinline__ unsigned lds_offset(const void* p) {
  return (unsigned)(size_t)((__attribute__((address_space(3))) const void*)p);
}

// ---- WMMA fragment loaders (ISA 7.12.2 layouts, wave32) --------------------
// A fragment, 16x32 f16, source row-major: p[m*ld + k]
__device__ __forceinline__ v16h frag_a_f16(const _Float16* p, int ld) {
  const int lane = lane_id();
  const int r    = lane & 15;
  const int kb   = (lane & 16) ? 8 : 0;
  const _Float16* row = p + r * ld + kb;
  v16h a;
#pragma unroll
  for (int i = 0; i < 8; ++i) {
    const int k = (i < 4) ? (2 * i) : (8 + 2 * i);  // 0,2,4,6,16,18,20,22
    a[2 * i]     = row[k];
    a[2 * i + 1] = row[k + 1];
  }
  return a;
}

// A fragment built from f32 source (converted to f16 on load)
__device__ __forceinline__ v16h frag_a_f32(const float* p, int ld) {
  const int lane = lane_id();
  const int r    = lane & 15;
  const int kb   = (lane & 16) ? 8 : 0;
  const float* row = p + r * ld + kb;
  v16h a;
#pragma unroll
  for (int i = 0; i < 8; ++i) {
    const int k = (i < 4) ? (2 * i) : (8 + 2 * i);
    a[2 * i]     = (_Float16)row[k];
    a[2 * i + 1] = (_Float16)row[k + 1];
  }
  return a;
}

// B fragment, 32x16, where memory holds B transposed: B(k,n) = p[n*ld + k]
// -> each lane reads two runs of 8 contiguous halves (k = kb..kb+7, kb+16..kb+23)
__device__ __forceinline__ v16h frag_b_cm(const _Float16* p, int ld) {
  const int lane = lane_id();
  const int n    = lane & 15;
  const int kb   = (lane & 16) ? 8 : 0;
  const _Float16* col = p + (size_t)n * ld + kb;
  v16h b;
#pragma unroll
  for (int i = 0; i < 16; ++i) b[i] = col[(i < 8) ? i : (i + 8)];
  return b;
}

// ---- Kernel 1: projection GEMM (q @ W + bias) -> f16, head layout ----------
// M=8192, N=1024, K=1024. Block: 128x128 tile, 256 thr (8 waves, each 64x32).
// VMODE==0: out[b][h][s][d]   (Q, K)
// VMODE==1: out[b][h][d][s]   (V transposed, for contiguous P*V B-fragments)
template <int VMODE>
__global__ __launch_bounds__(256)
void proj_gemm_k(const float* __restrict__ A, const float* __restrict__ W,
                 const float* __restrict__ bias, _Float16* __restrict__ out) {
  __shared__ _Float16 sA[128][40];   // [m][k], padded
  __shared__ _Float16 sBT[128][32];  // [n][k]  (transposed -> contiguous frags)
  const int m0 = blockIdx.x * 128;
  const int n0 = blockIdx.y * 128;
  const int t = (int)threadIdx.x;
  const int wave = t >> 5, lane = t & 31;
  const int wm = (wave >> 2) * 64;   // 2 wave-rows
  const int wn = (wave & 3) * 32;    // 4 wave-cols
  v8f acc[4][2] = {};
  for (int k0 = 0; k0 < HIDDEN; k0 += 32) {
    {
      const int r = t >> 1, c = (t & 1) * 16;
      const float* g = A + (size_t)(m0 + r) * HIDDEN + k0 + c;
#pragma unroll
      for (int j = 0; j < 16; ++j) sA[r][c + j] = (_Float16)g[j];
    }
    {
      const int r = t >> 3, c = (t & 7) * 16;  // r: k row 0..31, c: n col
      const float* g = W + (size_t)(k0 + r) * HIDDEN + n0 + c;
#pragma unroll
      for (int j = 0; j < 16; ++j) sBT[c + j][r] = (_Float16)g[j];
    }
    __syncthreads();
    v16h b0 = frag_b_cm(&sBT[wn][0], 32);
    v16h b1 = frag_b_cm(&sBT[wn + 16][0], 32);
#pragma unroll
    for (int mi = 0; mi < 4; ++mi) {
      v16h a = frag_a_f16(&sA[wm + mi * 16][0], 40);
      acc[mi][0] = WMMA_F16(a, b0, acc[mi][0]);
      acc[mi][1] = WMMA_F16(a, b1, acc[mi][1]);
    }
    __syncthreads();
  }
  const int nl = lane & 15;
  const int rb = (lane & 16) ? 8 : 0;
#pragma unroll
  for (int mi = 0; mi < 4; ++mi) {
#pragma unroll
    for (int ni = 0; ni < 2; ++ni) {
      const int ncol = n0 + wn + ni * 16 + nl;
      const float bvv = bias[ncol];
      const int h = ncol >> 6, d = ncol & 63;
#pragma unroll
      for (int i = 0; i < 8; ++i) {
        const int m = m0 + wm + mi * 16 + rb + i;
        const int b = m >> 11, s = m & (SEQ - 1);
        const _Float16 val = (_Float16)(acc[mi][ni][i] + bvv);
        if (VMODE)
          out[((size_t)((b * NUM_HEADS + h) * HEAD_DIM + d)) * SEQ + s] = val;
        else
          out[((size_t)((b * NUM_HEADS + h) * SEQ + s)) * HEAD_DIM + d] = val;
      }
    }
  }
}

// ---- Kernel 2: attention. One block = (b, h, 32-query tile). 256 thr. ------
// Qh/Kh: [b][h][S][64] f16 ; Vt: [b][h][64][S] f16 (transposed)
__global__ __launch_bounds__(256)
void attn_k(const _Float16* __restrict__ Qh, const _Float16* __restrict__ Kh,
            const _Float16* __restrict__ Vt, const float* __restrict__ mask,
            float* __restrict__ aw, float* __restrict__ ctx) {
  extern __shared__ float sc[];                   // [32][2048] scores (256 KB)
  __shared__ _Float16 skt[8][2][16 * HEAD_DIM];   // per-wave double buf (32 KB)
  __shared__ float red[32][8];
  __shared__ float rowmax[32];
  __shared__ float rinv[32];
  const int qt = (int)blockIdx.x;      // 32-row query tile, 0..63
  const int h  = (int)blockIdx.y;
  const int b  = (int)blockIdx.z;
  const int t = (int)threadIdx.x;
  const int wave = t >> 5, lane = t & 31;

  const size_t headoff = (size_t)(b * NUM_HEADS + h) * SEQ * HEAD_DIM;
  const _Float16* Qb = Qh + headoff + (size_t)qt * 32 * HEAD_DIM;  // [32][64]
  const _Float16* Kb = Kh + headoff;                               // [2048][64]
  const _Float16* Vb = Vt + headoff;                               // [64][2048]
  const float* mrow = mask + (size_t)b * SEQ;

  // Q fragments: 2 m-tiles x 2 k-steps
  v16h qa[2][2];
  qa[0][0] = frag_a_f16(Qb, HEAD_DIM);
  qa[0][1] = frag_a_f16(Qb + 32, HEAD_DIM);
  qa[1][0] = frag_a_f16(Qb + 16 * HEAD_DIM, HEAD_DIM);
  qa[1][1] = frag_a_f16(Qb + 16 * HEAD_DIM + 32, HEAD_DIM);

  const int n  = lane & 15;
  const int rb = (lane & 16) ? 8 : 0;

  // ---- scores = Q K^T / 8 + mask*(-1e9)  -> LDS ----
#if USE_ASYNC_K
  // Double-buffered async staging: DMA of tile i+1 overlaps WMMA of tile i.
  const unsigned sktBase = lds_offset(&skt[wave][0][0]);
  const unsigned dstLane = sktBase + (unsigned)lane * 16u;
#define ISSUE_K_TILE(KT, BUF)                                                  \
  {                                                                            \
    const _Float16* Kt_ = Kb + (size_t)(KT) * 16 * HEAD_DIM;                   \
    _Pragma("unroll") for (int i_ = 0; i_ < 4; ++i_) {                         \
      const unsigned ldst_ = dstLane + (unsigned)(BUF) * 2048u +               \
                             (unsigned)i_ * 512u;                              \
      const unsigned long long gsrc_ =                                         \
          (unsigned long long)(size_t)(Kt_ + (lane + 32 * i_) * 8);            \
      asm volatile("global_load_async_to_lds_b128 %0, %1, off" ::"v"(ldst_),   \
                   "v"(gsrc_)                                                  \
                   : "memory");                                                \
    }                                                                          \
  }
  ISSUE_K_TILE(wave, 0)
  for (int it = 0; it < 16; ++it) {
    const int kt = wave + 8 * it;
    if (it < 15) {
      ISSUE_K_TILE(kt + 8, (it + 1) & 1)
      // in-order completion: <=4 outstanding => current tile has landed
      asm volatile("s_wait_asynccnt 0x4" ::: "memory");
    } else {
      asm volatile("s_wait_asynccnt 0x0" ::: "memory");
    }
    const _Float16* Ktile = &skt[wave][it & 1][0];
    v16h kb0 = frag_b_cm(Ktile, HEAD_DIM);       // contraction dims 0..31
    v16h kb1 = frag_b_cm(Ktile + 32, HEAD_DIM);  // dims 32..63
    const float mval = mrow[kt * 16 + n] * (-1e9f);
#pragma unroll
    for (int mi = 0; mi < 2; ++mi) {
      v8f s = {};
      s = WMMA_F16(qa[mi][0], kb0, s);
      s = WMMA_F16(qa[mi][1], kb1, s);
#pragma unroll
      for (int i = 0; i < 8; ++i)
        sc[(size_t)(mi * 16 + rb + i) * SEQ + kt * 16 + n] =
            s[i] * 0.125f + mval;
    }
  }
#undef ISSUE_K_TILE
#else
  for (int kt = wave; kt < SEQ / 16; kt += 8) {
    const _Float16* Kt = Kb + (size_t)kt * 16 * HEAD_DIM;
    v16h kb0 = frag_b_cm(Kt, HEAD_DIM);
    v16h kb1 = frag_b_cm(Kt + 32, HEAD_DIM);
    const float mval = mrow[kt * 16 + n] * (-1e9f);
#pragma unroll
    for (int mi = 0; mi < 2; ++mi) {
      v8f s = {};
      s = WMMA_F16(qa[mi][0], kb0, s);
      s = WMMA_F16(qa[mi][1], kb1, s);
#pragma unroll
      for (int i = 0; i < 8; ++i)
        sc[(size_t)(mi * 16 + rb + i) * SEQ + kt * 16 + n] =
            s[i] * 0.125f + mval;
    }
  }
#endif
  __syncthreads();

  // ---- softmax over each of 32 rows (len 2048); 8 threads per row ----
  const int r = t & 31, slot = t >> 5;
  const int c0 = slot * 256;
  float mx = -3.4e38f;
  for (int j = 0; j < 256; ++j) mx = fmaxf(mx, sc[(size_t)r * SEQ + c0 + j]);
  red[r][slot] = mx;
  __syncthreads();
  if (t < 32) {
    float m2 = red[t][0];
#pragma unroll
    for (int j = 1; j < 8; ++j) m2 = fmaxf(m2, red[t][j]);
    rowmax[t] = m2;
  }
  __syncthreads();
  const float rm = rowmax[r];
  float sum = 0.f;
  for (int j = 0; j < 256; ++j) {
    const size_t idx = (size_t)r * SEQ + c0 + j;
    const float e = __expf(sc[idx] - rm);
    sc[idx] = e;
    sum += e;
  }
  red[r][slot] = sum;
  __syncthreads();
  if (t < 32) {
    float s2 = red[t][0];
#pragma unroll
    for (int j = 1; j < 8; ++j) s2 += red[t][j];
    rinv[t] = 1.0f / s2;
  }
  __syncthreads();
  // normalize in LDS + coalesced store of attention_weights (written once)
  float* awb = aw + ((size_t)((b * NUM_HEADS + h) * SEQ + qt * 32)) * SEQ;
  for (int idx = t; idx < 32 * SEQ; idx += 256) {
    const int rr = idx >> 11;
    const float p = sc[idx] * rinv[rr];
    sc[idx] = p;
    awb[idx] = p;
  }
  __syncthreads();

  // ---- context = P @ V ; 8 waves = 2 m-tiles x 4 d-slices ----
  const int mi = wave >> 2;        // 0..1
  const int ns = wave & 3;         // 0..3 (16-wide slice of d)
  v8f cacc = {};
  for (int kk = 0; kk < SEQ; kk += 32) {
    v16h pa = frag_a_f32(sc + (size_t)mi * 16 * SEQ + kk, SEQ);   // P (LDS)
    v16h vb = frag_b_cm(Vb + (size_t)(ns * 16) * SEQ + kk, SEQ);  // V^T (glob)
    cacc = WMMA_F16(pa, vb, cacc);
  }
  float* cb = ctx + ((size_t)(b * SEQ + qt * 32 + mi * 16)) * HIDDEN +
              h * HEAD_DIM + ns * 16 + n;
#pragma unroll
  for (int i = 0; i < 8; ++i) cb[(size_t)(rb + i) * HIDDEN] = cacc[i];
}

// ---- Kernel 3: output GEMM  out[m][n] = ctx @ Wo + bo (f32 out) ------------
__global__ __launch_bounds__(256)
void out_gemm_k(const float* __restrict__ A, const float* __restrict__ W,
                const float* __restrict__ bias, float* __restrict__ out) {
  __shared__ _Float16 sA[128][40];
  __shared__ _Float16 sBT[128][32];
  const int m0 = blockIdx.x * 128;
  const int n0 = blockIdx.y * 128;
  const int t = (int)threadIdx.x;
  const int wave = t >> 5, lane = t & 31;
  const int wm = (wave >> 2) * 64;
  const int wn = (wave & 3) * 32;
  v8f acc[4][2] = {};
  for (int k0 = 0; k0 < HIDDEN; k0 += 32) {
    {
      const int r = t >> 1, c = (t & 1) * 16;
      const float* g = A + (size_t)(m0 + r) * HIDDEN + k0 + c;
#pragma unroll
      for (int j = 0; j < 16; ++j) sA[r][c + j] = (_Float16)g[j];
    }
    {
      const int r = t >> 3, c = (t & 7) * 16;
      const float* g = W + (size_t)(k0 + r) * HIDDEN + n0 + c;
#pragma unroll
      for (int j = 0; j < 16; ++j) sBT[c + j][r] = (_Float16)g[j];
    }
    __syncthreads();
    v16h b0 = frag_b_cm(&sBT[wn][0], 32);
    v16h b1 = frag_b_cm(&sBT[wn + 16][0], 32);
#pragma unroll
    for (int mi = 0; mi < 4; ++mi) {
      v16h a = frag_a_f16(&sA[wm + mi * 16][0], 40);
      acc[mi][0] = WMMA_F16(a, b0, acc[mi][0]);
      acc[mi][1] = WMMA_F16(a, b1, acc[mi][1]);
    }
    __syncthreads();
  }
  const int nl = lane & 15;
  const int rb = (lane & 16) ? 8 : 0;
#pragma unroll
  for (int mi = 0; mi < 4; ++mi) {
#pragma unroll
    for (int ni = 0; ni < 2; ++ni) {
      const int ncol = n0 + wn + ni * 16 + nl;
      const float bvv = bias[ncol];
#pragma unroll
      for (int i = 0; i < 8; ++i) {
        const int m = m0 + wm + mi * 16 + rb + i;
        out[(size_t)m * HIDDEN + ncol] = acc[mi][ni][i] + bvv;
      }
    }
  }
}

extern "C" void kernel_launch(void* const* d_in, const int* in_sizes, int n_in,
                              void* d_out, int out_size, void* d_ws, size_t ws_size,
                              hipStream_t stream) {
  const float* q    = (const float*)d_in[0];
  const float* mask = (const float*)d_in[1];
  const float* Wq   = (const float*)d_in[2];
  const float* bq   = (const float*)d_in[3];
  const float* Wk   = (const float*)d_in[4];
  const float* bk   = (const float*)d_in[5];
  const float* Wv   = (const float*)d_in[6];
  const float* bv   = (const float*)d_in[7];
  const float* Wo   = (const float*)d_in[8];
  const float* bo   = (const float*)d_in[9];

  float* out = (float*)d_out;
  float* ctx_out = out;                                   // [B][S][H] f32
  float* aw_out  = out + (size_t)BATCH * SEQ * HIDDEN;    // [B][nH][S][S] f32

  // Workspace: Qh/Kh/Vt f16 (16 MB each) + ctx f32 (32 MB) = 80 MB
  char* ws = (char*)d_ws;
  _Float16* Qh = (_Float16*)(ws);
  _Float16* Kh = (_Float16*)(ws + ((size_t)16 << 20));
  _Float16* Vt = (_Float16*)(ws + ((size_t)32 << 20));
  float*    ctx = (float*)(ws + ((size_t)48 << 20));

  dim3 blk(256);
  dim3 g1(64, 8);   // M/128 x N/128
  proj_gemm_k<0><<<g1, blk, 0, stream>>>(q, Wq, bq, Qh);
  proj_gemm_k<0><<<g1, blk, 0, stream>>>(q, Wk, bk, Kh);
  proj_gemm_k<1><<<g1, blk, 0, stream>>>(q, Wv, bv, Vt);

  const int attn_lds = 32 * SEQ * (int)sizeof(float);     // 256 KB dynamic LDS
  (void)hipFuncSetAttribute((const void*)attn_k,
                            hipFuncAttributeMaxDynamicSharedMemorySize, attn_lds);
  dim3 g2(SEQ / 32, NUM_HEADS, BATCH);
  attn_k<<<g2, blk, attn_lds, stream>>>(Qh, Kh, Vt, mask, aw_out, ctx);

  dim3 g3(64, 8);
  out_gemm_k<<<g3, blk, 0, stream>>>(ctx, Wo, bo, ctx_out);
}
